// TensorProductScoreModelV6_26431228740212
// MI455X (gfx1250) — matrix operand order repaired and checked by reading, not compiled, not run on hardware
//
#include <hip/hip_runtime.h>
#include <hip/hip_bf16.h>

typedef __attribute__((ext_vector_type(16))) _Float16 v16h;
typedef __attribute__((ext_vector_type(8)))  float    v8f;

#if defined(__HIP_DEVICE_COMPILE__)
#define WMMA_F16(A, B, C) __builtin_amdgcn_wmma_f32_16x16x32_f16(false, (A), false, (B), (short)0, (C), false, false)
#else
#define WMMA_F16(A, B, C) (C)
#endif

#define NSx   16
#define NVx   4
#define KDIM  48
#define W1N   320
#define W2N   416
#define W1PAD 324   // 320 + 4 pad: row stride -> 4-bank offset per row, conflict-free for 16 lanes
#define W2PAD 420   // 416 + 4 pad: 36-bank offset per row, all 16 rows distinct banks
#define WPB   2     // waves (16-edge tiles) per block

// ---------------------------------------------------------------------------
// Repack a KDIM x Ncols f32 weight matrix into per-lane f16 WMMA B fragments.
// Fragment layout used here (16-bit B, 16x16x32): VGPR j, lanes 0-15 hold rows K=2j,2j+1
// (cols = lane), lanes 16-31 hold rows K=16+2j,16+2j+1. Two 32-row k-chunks.
// dst index p = ((tile*2 + chunk)*32 + lane)*16 + elem
// ---------------------------------------------------------------------------
__global__ void pack_weights_kernel(const float* __restrict__ w, int Kdim, int Ncols,
                                    int ntiles, _Float16* __restrict__ dst) {
    int p = blockIdx.x * blockDim.x + threadIdx.x;
    int total = ntiles * 2 * 32 * 16;
    if (p >= total) return;
    int elem  = p & 15;
    int lane  = (p >> 4) & 31;
    int chunk = (p >> 9) & 1;
    int tile  = p >> 10;
    int col   = tile * 16 + (lane & 15);
    int j     = elem >> 1;
    int kloc  = 2 * j + (elem & 1) + (lane >> 4) * 16;
    int K     = chunk * 32 + kloc;
    float v = (K < Kdim && col < Ncols) ? w[(size_t)K * Ncols + col] : 0.0f;
    dst[p] = (_Float16)v;
}

// ---------------------------------------------------------------------------
// node0[i][k] = sum_c emb_tables[c][x_cat[i][c]][k] + x_scalar[i] @ node_w + node_b
// ---------------------------------------------------------------------------
__global__ void node_init_kernel(const int* __restrict__ x_cat, const float* __restrict__ x_scalar,
                                 const float* __restrict__ emb, const float* __restrict__ nw,
                                 const float* __restrict__ nb, float* __restrict__ node0, int n) {
    __shared__ float s_nw[32 * 16];
    __shared__ float s_nb[16];
    for (int i = threadIdx.x; i < 512; i += blockDim.x) s_nw[i] = nw[i];
    if (threadIdx.x < 16) s_nb[threadIdx.x] = nb[threadIdx.x];
    __syncthreads();
    int i = blockIdx.x * blockDim.x + threadIdx.x;
    if (i >= n) return;
    float o[16];
#pragma unroll
    for (int k = 0; k < 16; ++k) o[k] = s_nb[k];
#pragma unroll
    for (int c = 0; c < 10; ++c) {
        int t = x_cat[i * 10 + c];
        const float* ep = emb + ((size_t)c * 119 + t) * 16;
#pragma unroll
        for (int k = 0; k < 16; ++k) o[k] += ep[k];
    }
    for (int j = 0; j < 32; ++j) {
        float xv = x_scalar[(size_t)i * 32 + j];
#pragma unroll
        for (int k = 0; k < 16; ++k) o[k] = fmaf(xv, s_nw[j * 16 + k], o[k]);
    }
#pragma unroll
    for (int k = 0; k < 16; ++k) node0[(size_t)i * 16 + k] = o[k];
}

// ---------------------------------------------------------------------------
// Per-edge geometry + RBF + edge-embedding MLP (42 -> 16 relu -> 16)
// ---------------------------------------------------------------------------
__global__ void edge_kernel(const float* __restrict__ pos, const int* __restrict__ ei,
                            const float* __restrict__ bond,
                            const float* __restrict__ ew1, const float* __restrict__ eb1,
                            const float* __restrict__ ew2, const float* __restrict__ eb2,
                            float* __restrict__ edge_emb, float* __restrict__ sh1o, int E_) {
    __shared__ float s_w1[42 * 16];
    __shared__ float s_b1[16];
    __shared__ float s_w2[16 * 16];
    __shared__ float s_b2[16];
    for (int i = threadIdx.x; i < 42 * 16; i += blockDim.x) s_w1[i] = ew1[i];
    for (int i = threadIdx.x; i < 256; i += blockDim.x) s_w2[i] = ew2[i];
    if (threadIdx.x < 16) { s_b1[threadIdx.x] = eb1[threadIdx.x]; s_b2[threadIdx.x] = eb2[threadIdx.x]; }
    __syncthreads();
    int e = blockIdx.x * blockDim.x + threadIdx.x;
    if (e >= E_) return;
    int s = ei[e], d = ei[E_ + e];
    float vx = pos[d * 3 + 0] - pos[s * 3 + 0];
    float vy = pos[d * 3 + 1] - pos[s * 3 + 1];
    float vz = pos[d * 3 + 2] - pos[s * 3 + 2];
    float dist = sqrtf(vx * vx + vy * vy + vz * vz + 1e-12f);
    float inv = 1.0f / (dist + 1e-8f);
    const float SQ3 = 1.7320508075688772f;
    sh1o[(size_t)e * 4 + 0] = SQ3 * vx * inv;
    sh1o[(size_t)e * 4 + 1] = SQ3 * vy * inv;
    sh1o[(size_t)e * 4 + 2] = SQ3 * vz * inv;
    float ein[42];
#pragma unroll
    for (int c = 0; c < 10; ++c) ein[c] = bond[(size_t)e * 10 + c];
    const float step = 5.0f / 31.0f;
    const float coeff = -0.5f / (step * step);
#pragma unroll
    for (int j = 0; j < 32; ++j) {
        float dd = dist - step * (float)j;
        ein[10 + j] = __expf(coeff * dd * dd);
    }
    float h[16];
#pragma unroll
    for (int u = 0; u < 16; ++u) h[u] = s_b1[u];
    for (int c = 0; c < 42; ++c) {
        float x = ein[c];
#pragma unroll
        for (int u = 0; u < 16; ++u) h[u] = fmaf(x, s_w1[c * 16 + u], h[u]);
    }
#pragma unroll
    for (int u = 0; u < 16; ++u) h[u] = fmaxf(h[u], 0.0f);
    float o[16];
#pragma unroll
    for (int k = 0; k < 16; ++k) o[k] = s_b2[k];
#pragma unroll
    for (int u = 0; u < 16; ++u) {
        float hv = h[u];
#pragma unroll
        for (int k = 0; k < 16; ++k) o[k] = fmaf(hv, s_w2[u * 16 + k], o[k]);
    }
#pragma unroll
    for (int k = 0; k < 16; ++k) edge_emb[(size_t)e * 16 + k] = o[k];
}

__global__ void count_kernel(const int* __restrict__ ei, int* __restrict__ cnt, int E_) {
    int e = blockIdx.x * blockDim.x + threadIdx.x;
    if (e < E_) atomicAdd(&cnt[ei[e]], 1);
}

// Build f16 A fragments (16x32 layout) for a 48-wide row held in LDS.
// lane<16 covers K in {0..15}(+base), lane>=16 covers {16..31}.
__device__ __forceinline__ void build_a_frags(const float* rp, int part, v16h& a0, v16h& a1) {
#pragma unroll
    for (int q = 0; q < 16; ++q) {
        int k = q + part * 16;
        a0[q] = (_Float16)rp[k];
        int K1 = 32 + k;
        a1[q] = (K1 < KDIM) ? (_Float16)rp[K1] : (_Float16)0.0f;
    }
}

// ---------------------------------------------------------------------------
// conv1: ea(16x48) -> relu(ea@W1+b1)(16x48) -> (@W2+b2)(16x320) -> per-edge
// matvecs with s_dst, scatter-sum via atomics.
// ---------------------------------------------------------------------------
__global__ __launch_bounds__(64) void conv1_kernel(
    const float* __restrict__ edge_emb, const float* __restrict__ node0,
    const int* __restrict__ ei, const float* __restrict__ sh1g,
    const _Float16* __restrict__ w1pk, const _Float16* __restrict__ w2pk,
    const float* __restrict__ b1, const float* __restrict__ b2,
    float* __restrict__ sum1, int E_) {
    __shared__ float s_ea[WPB][16][48];
    __shared__ float s_w[WPB][16][W1PAD];
    int wv = threadIdx.x >> 5, lane = threadIdx.x & 31;
    int tile = blockIdx.x * WPB + wv;
    int e0 = tile * 16;
    int el = lane & 15, part = lane >> 4;

    { // stage ea = [edge_emb(16) | node0[src](16) | node0[dst](16)]
        int e = e0 + el;
        bool ok = e < E_;
        int s = ok ? ei[e] : 0;
        int d = ok ? ei[E_ + e] : 0;
        for (int c0 = 0; c0 < 24; ++c0) {
            int c = part * 24 + c0;
            float x = 0.0f;
            if (ok) {
                if (c < 16) x = edge_emb[(size_t)e * 16 + c];
                else if (c < 32) x = node0[(size_t)s * 16 + (c - 16)];
                else x = node0[(size_t)d * 16 + (c - 32)];
            }
            s_ea[wv][el][c] = x;
        }
    }
    __syncthreads();

    v16h a0, a1;
    build_a_frags(&s_ea[wv][lane & 15][0], part, a0, a1);

    const v16h* B1 = (const v16h*)w1pk;
    const v16h* B2 = (const v16h*)w2pk;

    // GEMM1: h = relu(ea@W1 + b1), stored into s_w[:,0:48]
#pragma unroll
    for (int t = 0; t < 3; ++t) {
        v8f acc;
        float bv = b1[t * 16 + (lane & 15)];
#pragma unroll
        for (int r = 0; r < 8; ++r) acc[r] = bv;
        v16h bf0 = B1[(t * 2 + 0) * 32 + lane];
        v16h bf1 = B1[(t * 2 + 1) * 32 + lane];
        acc = WMMA_F16(a0, bf0, acc);
        acc = WMMA_F16(a1, bf1, acc);
#pragma unroll
        for (int r = 0; r < 8; ++r)
            s_w[wv][r + part * 8][t * 16 + (lane & 15)] = fmaxf(acc[r], 0.0f);
    }
    __syncthreads();

    v16h h0, h1;
    build_a_frags(&s_w[wv][lane & 15][0], part, h0, h1);
    __syncthreads();

    // GEMM2: w = h@W2 + b2 (16x320) into padded LDS
#pragma unroll
    for (int t = 0; t < 20; ++t) {
        v8f acc;
        float bv = b2[t * 16 + (lane & 15)];
#pragma unroll
        for (int r = 0; r < 8; ++r) acc[r] = bv;
        v16h bf0 = B2[(t * 2 + 0) * 32 + lane];
        v16h bf1 = B2[(t * 2 + 1) * 32 + lane];
        acc = WMMA_F16(h0, bf0, acc);
        acc = WMMA_F16(h1, bf1, acc);
#pragma unroll
        for (int r = 0; r < 8; ++r)
            s_w[wv][r + part * 8][t * 16 + (lane & 15)] = acc[r];
    }
    __syncthreads();

    // consume: one lane per edge
    if (part == 0) {
        int e = e0 + el;
        if (e < E_) {
            int s = ei[e];
            const float* wr = &s_w[wv][el][0];
            float sd[16];
#pragma unroll
            for (int u = 0; u < 16; ++u) sd[u] = s_ea[wv][el][32 + u];
            float o0[16];
#pragma unroll
            for (int k = 0; k < 16; ++k) o0[k] = 0.0f;
#pragma unroll
            for (int u = 0; u < 16; ++u) {
                float su = sd[u];
#pragma unroll
                for (int k = 0; k < 16; ++k) o0[k] = fmaf(wr[u * 16 + k], su, o0[k]);
            }
            float ob[4] = {0.f, 0.f, 0.f, 0.f};
#pragma unroll
            for (int u = 0; u < 16; ++u) {
                float su = sd[u];
#pragma unroll
                for (int q = 0; q < 4; ++q) ob[q] = fmaf(wr[256 + u * 4 + q], su, ob[q]);
            }
            const float aa = 0.25f; // 1/sqrt(ns)
            float sx = sh1g[(size_t)e * 4 + 0];
            float sy = sh1g[(size_t)e * 4 + 1];
            float sz = sh1g[(size_t)e * 4 + 2];
            float* dstp = sum1 + (size_t)s * 28;
#pragma unroll
            for (int k = 0; k < 16; ++k) atomicAdd(dstp + k, o0[k] * aa);
#pragma unroll
            for (int q = 0; q < 4; ++q) {
                float b = ob[q] * aa;
                atomicAdd(dstp + 16 + q * 3 + 0, b * sx);
                atomicAdd(dstp + 16 + q * 3 + 1, b * sy);
                atomicAdd(dstp + 16 + q * 3 + 2, b * sz);
            }
        }
    }
}

// ---------------------------------------------------------------------------
// conv2: ea(16x48) -> relu(@W1) -> (@W2)(16x416) -> equivariant consume
// ---------------------------------------------------------------------------
__global__ __launch_bounds__(64) void conv2_kernel(
    const float* __restrict__ edge_emb, const float* __restrict__ node1,
    const int* __restrict__ ei, const float* __restrict__ sh1g,
    const _Float16* __restrict__ w1pk, const _Float16* __restrict__ w2pk,
    const float* __restrict__ b1, const float* __restrict__ b2,
    float* __restrict__ sum2, int E_) {
    __shared__ float s_ea[WPB][16][48];
    __shared__ float s_v[WPB][16][12];
    __shared__ float s_w[WPB][16][W2PAD];
    int wv = threadIdx.x >> 5, lane = threadIdx.x & 31;
    int tile = blockIdx.x * WPB + wv;
    int e0 = tile * 16;
    int el = lane & 15, part = lane >> 4;

    { // stage ea (48) + v (12 = node1[dst][16:28])
        int e = e0 + el;
        bool ok = e < E_;
        int s = ok ? ei[e] : 0;
        int d = ok ? ei[E_ + e] : 0;
        for (int c0 = 0; c0 < 30; ++c0) {
            int c = part * 30 + c0;
            float x = 0.0f;
            if (ok) {
                if (c < 16) x = edge_emb[(size_t)e * 16 + c];
                else if (c < 32) x = node1[(size_t)s * 28 + (c - 16)];
                else if (c < 48) x = node1[(size_t)d * 28 + (c - 32)];
                else x = node1[(size_t)d * 28 + 16 + (c - 48)];
            }
            if (c < 48) s_ea[wv][el][c] = x;
            else s_v[wv][el][c - 48] = x;
        }
    }
    __syncthreads();

    v16h a0, a1;
    build_a_frags(&s_ea[wv][lane & 15][0], part, a0, a1);

    const v16h* B1 = (const v16h*)w1pk;
    const v16h* B2 = (const v16h*)w2pk;

#pragma unroll
    for (int t = 0; t < 3; ++t) {
        v8f acc;
        float bv = b1[t * 16 + (lane & 15)];
#pragma unroll
        for (int r = 0; r < 8; ++r) acc[r] = bv;
        v16h bf0 = B1[(t * 2 + 0) * 32 + lane];
        v16h bf1 = B1[(t * 2 + 1) * 32 + lane];
        acc = WMMA_F16(a0, bf0, acc);
        acc = WMMA_F16(a1, bf1, acc);
#pragma unroll
        for (int r = 0; r < 8; ++r)
            s_w[wv][r + part * 8][t * 16 + (lane & 15)] = fmaxf(acc[r], 0.0f);
    }
    __syncthreads();

    v16h h0, h1;
    build_a_frags(&s_w[wv][lane & 15][0], part, h0, h1);
    __syncthreads();

#pragma unroll
    for (int t = 0; t < 26; ++t) {
        v8f acc;
        float bv = b2[t * 16 + (lane & 15)];
#pragma unroll
        for (int r = 0; r < 8; ++r) acc[r] = bv;
        v16h bf0 = B2[(t * 2 + 0) * 32 + lane];
        v16h bf1 = B2[(t * 2 + 1) * 32 + lane];
        acc = WMMA_F16(h0, bf0, acc);
        acc = WMMA_F16(h1, bf1, acc);
#pragma unroll
        for (int r = 0; r < 8; ++r)
            s_w[wv][r + part * 8][t * 16 + (lane & 15)] = acc[r];
    }
    __syncthreads();

    if (part == 0) {
        int e = e0 + el;
        if (e < E_) {
            int sidx = ei[e];
            const float* wr = &s_w[wv][el][0];
            float s[16];
#pragma unroll
            for (int u = 0; u < 16; ++u) s[u] = s_ea[wv][el][32 + u];
            float vv[4][3];
#pragma unroll
            for (int u = 0; u < 4; ++u)
#pragma unroll
                for (int i = 0; i < 3; ++i) vv[u][i] = s_v[wv][el][u * 3 + i];
            float sx = sh1g[(size_t)e * 4 + 0];
            float sy = sh1g[(size_t)e * 4 + 1];
            float sz = sh1g[(size_t)e * 4 + 2];
            float vd[4];
#pragma unroll
            for (int u = 0; u < 4; ++u)
                vd[u] = vv[u][0] * sx + vv[u][1] * sy + vv[u][2] * sz;

            const float A0 = 0.22360679774997896f;      // 1/sqrt(ns+nv)=1/sqrt(20)
            const float INV_SQ3 = 0.5773502691896258f;
            const float AE = 0.35355339059327373f;      // (1/sqrt(nv))/sqrt(2) = 0.5/sqrt2

            float o0[16];
#pragma unroll
            for (int k = 0; k < 16; ++k) o0[k] = 0.0f;
#pragma unroll
            for (int u = 0; u < 16; ++u) {
                float su = s[u];
#pragma unroll
                for (int k = 0; k < 16; ++k) o0[k] = fmaf(wr[u * 16 + k], su, o0[k]);
            }
#pragma unroll
            for (int u = 0; u < 4; ++u) {
                float vu = vd[u] * INV_SQ3;
#pragma unroll
                for (int k = 0; k < 16; ++k) o0[k] = fmaf(wr[256 + u * 16 + k], vu, o0[k]);
            }
            float oc[4] = {0.f, 0.f, 0.f, 0.f};
#pragma unroll
            for (int u = 0; u < 16; ++u) {
                float su = s[u];
#pragma unroll
                for (int q = 0; q < 4; ++q) oc[q] = fmaf(wr[320 + u * 4 + q], su, oc[q]);
            }
            float o1[4][3];
#pragma unroll
            for (int q = 0; q < 4; ++q) {
                o1[q][0] = oc[q] * sx; o1[q][1] = oc[q] * sy; o1[q][2] = oc[q] * sz;
            }
#pragma unroll
            for (int u = 0; u < 4; ++u) {
#pragma unroll
                for (int q = 0; q < 4; ++q) {
                    float dw = wr[384 + u * 4 + q];
#pragma unroll
                    for (int i = 0; i < 3; ++i) o1[q][i] = fmaf(dw, vv[u][i], o1[q][i]);
                }
            }
            float cv[4][3];
#pragma unroll
            for (int u = 0; u < 4; ++u) {
                cv[u][0] = vv[u][1] * sz - vv[u][2] * sy;
                cv[u][1] = vv[u][2] * sx - vv[u][0] * sz;
                cv[u][2] = vv[u][0] * sy - vv[u][1] * sx;
            }
            float oe[4][3];
#pragma unroll
            for (int q = 0; q < 4; ++q)
#pragma unroll
                for (int i = 0; i < 3; ++i) oe[q][i] = 0.0f;
#pragma unroll
            for (int u = 0; u < 4; ++u) {
#pragma unroll
                for (int q = 0; q < 4; ++q) {
                    float ww = wr[400 + u * 4 + q];
#pragma unroll
                    for (int i = 0; i < 3; ++i) oe[q][i] = fmaf(ww, cv[u][i], oe[q][i]);
                }
            }
            float* dstp = sum2 + (size_t)sidx * 40;
#pragma unroll
            for (int k = 0; k < 16; ++k) atomicAdd(dstp + k, o0[k] * A0);
#pragma unroll
            for (int q = 0; q < 4; ++q)
#pragma unroll
                for (int i = 0; i < 3; ++i) atomicAdd(dstp + 16 + q * 3 + i, o1[q][i] * A0);
#pragma unroll
            for (int q = 0; q < 4; ++q)
#pragma unroll
                for (int i = 0; i < 3; ++i) atomicAdd(dstp + 28 + q * 3 + i, oe[q][i] * AE);
        }
    }
}

__global__ void finalize_kernel(const float* __restrict__ sums, const int* __restrict__ cnt,
                                float* __restrict__ out, int n, int C) {
    int i = blockIdx.x * blockDim.x + threadIdx.x;
    if (i >= n * C) return;
    int node = i / C;
    int c = cnt[node];
    float denom = (float)(c > 0 ? c : 1);
    out[i] = sums[i] / denom;
}

// ---------------------------------------------------------------------------
extern "C" void kernel_launch(void* const* d_in, const int* in_sizes, int n_in,
                              void* d_out, int out_size, void* d_ws, size_t ws_size,
                              hipStream_t stream) {
    const int*   x_cat    = (const int*)d_in[0];
    const float* x_scalar = (const float*)d_in[1];
    const float* pos      = (const float*)d_in[2];
    const int*   ei       = (const int*)d_in[3];
    const float* bond     = (const float*)d_in[4];
    const float* emb      = (const float*)d_in[5];
    const float* node_w   = (const float*)d_in[6];
    const float* node_b   = (const float*)d_in[7];
    const float* em_w1    = (const float*)d_in[8];
    const float* em_b1    = (const float*)d_in[9];
    const float* em_w2    = (const float*)d_in[10];
    const float* em_b2    = (const float*)d_in[11];
    const float* c1_w1    = (const float*)d_in[12];
    const float* c1_b1    = (const float*)d_in[13];
    const float* c1_w2    = (const float*)d_in[14];
    const float* c1_b2    = (const float*)d_in[15];
    const float* c2_w1    = (const float*)d_in[16];
    const float* c2_b1    = (const float*)d_in[17];
    const float* c2_w2    = (const float*)d_in[18];
    const float* c2_b2    = (const float*)d_in[19];

    const int N = in_sizes[0] / 10;
    const int E = in_sizes[3] / 2;

    // workspace carve-out (256B aligned)
    char* ws = (char*)d_ws;
    size_t off = 0;
    auto take = [&](size_t bytes) -> void* {
        void* p = ws + off;
        off = (off + bytes + 255) & ~(size_t)255;
        return p;
    };
    const int PK_C1W1 = 3 * 1024, PK_C1W2 = 20 * 1024, PK_C2W1 = 3 * 1024, PK_C2W2 = 26 * 1024;
    _Float16* pk_c1w1 = (_Float16*)take((size_t)PK_C1W1 * 2);
    _Float16* pk_c1w2 = (_Float16*)take((size_t)PK_C1W2 * 2);
    _Float16* pk_c2w1 = (_Float16*)take((size_t)PK_C2W1 * 2);
    _Float16* pk_c2w2 = (_Float16*)take((size_t)PK_C2W2 * 2);
    float* node0    = (float*)take((size_t)N * 16 * 4);
    float* edge_emb = (float*)take((size_t)E * 16 * 4);
    float* sh1      = (float*)take((size_t)E * 4 * 4);
    int*   cnt      = (int*)take((size_t)N * 4);
    float* sum1     = (float*)take((size_t)N * 28 * 4);
    float* node1    = (float*)take((size_t)N * 28 * 4);
    float* sum2     = (float*)take((size_t)N * 40 * 4);

    (void)hipMemsetAsync(cnt,  0, (size_t)N * 4, stream);
    (void)hipMemsetAsync(sum1, 0, (size_t)N * 28 * 4, stream);
    (void)hipMemsetAsync(sum2, 0, (size_t)N * 40 * 4, stream);

    // weight repack into WMMA B-fragment layout
    pack_weights_kernel<<<(PK_C1W1 + 255) / 256, 256, 0, stream>>>(c1_w1, KDIM, 48,  3,  pk_c1w1);
    pack_weights_kernel<<<(PK_C1W2 + 255) / 256, 256, 0, stream>>>(c1_w2, KDIM, W1N, 20, pk_c1w2);
    pack_weights_kernel<<<(PK_C2W1 + 255) / 256, 256, 0, stream>>>(c2_w1, KDIM, 48,  3,  pk_c2w1);
    pack_weights_kernel<<<(PK_C2W2 + 255) / 256, 256, 0, stream>>>(c2_w2, KDIM, W2N, 26, pk_c2w2);

    node_init_kernel<<<(N + 255) / 256, 256, 0, stream>>>(x_cat, x_scalar, emb, node_w, node_b, node0, N);
    edge_kernel<<<(E + 255) / 256, 256, 0, stream>>>(pos, ei, bond, em_w1, em_b1, em_w2, em_b2,
                                                     edge_emb, sh1, E);
    count_kernel<<<(E + 255) / 256, 256, 0, stream>>>(ei, cnt, E);

    int etiles = (E + 15) / 16;
    int cblocks = (etiles + WPB - 1) / WPB;
    conv1_kernel<<<cblocks, 32 * WPB, 0, stream>>>(edge_emb, node0, ei, sh1,
                                                   pk_c1w1, pk_c1w2, c1_b1, c1_b2, sum1, E);
    finalize_kernel<<<(N * 28 + 255) / 256, 256, 0, stream>>>(sum1, cnt, node1, N, 28);
    conv2_kernel<<<cblocks, 32 * WPB, 0, stream>>>(edge_emb, node1, ei, sh1,
                                                   pk_c2w1, pk_c2w2, c2_b1, c2_b2, sum2, E);
    finalize_kernel<<<(N * 40 + 255) / 256, 256, 0, stream>>>(sum2, cnt, (float*)d_out, N, 40);
}
